// MultiHeadDiffAttention_41180146434598
// MI455X (gfx1250) — compile-verified, matrix operand order checked by reading, not run on hardware
//
#include <hip/hip_runtime.h>

// ---------------------------------------------------------------------------
// Problem dims (fixed by reference)
// ---------------------------------------------------------------------------
#define B_   2
#define T_   2048
#define D_   512
#define H_   8
#define DH_  64
#define M_   512
#define HD2_ 1024  // 2*H*DH

#define LAMBDA_INIT_F 0.4707130183432603f   // 0.8 - 0.6*exp(-0.6)
#define SCALE_F 0.125f                      // 1/sqrt(64)
#define EPS_F 1e-6f

// ---------------------------------------------------------------------------
// CDNA5 WMMA plumbing (wave32, v_wmma_f32_16x16x32_bf16)
// ---------------------------------------------------------------------------
typedef __attribute__((ext_vector_type(16))) __bf16        bf16x16;
typedef __attribute__((ext_vector_type(8)))  float         f32x8;
typedef __attribute__((ext_vector_type(4)))  unsigned int  u32x4;

union FragU { bf16x16 v; u32x4 q[2]; };

__device__ __forceinline__ unsigned short f2bf(float f) {
    unsigned int u = __float_as_uint(f);
    u += 0x7FFFu + ((u >> 16) & 1u);           // round-to-nearest-even
    return (unsigned short)(u >> 16);
}

// Per-lane 16B gather from global directly into LDS (ASYNCcnt-tracked).
// Generic pointer low 32 bits == LDS byte offset (ISA 10.2 aperture rule).
__device__ __forceinline__ void async_copy16(const void* g, void* l) {
    asm volatile("global_load_async_to_lds_b128 %0, %1, off"
                 :: "v"((unsigned int)(unsigned long long)l),
                    "v"((unsigned long long)g)
                 : "memory");
}
__device__ __forceinline__ void async_wait() {
    asm volatile("s_wait_asynccnt 0x0" ::: "memory");
}

// Butterfly reductions across the 16-lane half that owns one C/D row,
// using ds_swizzle group-of-32 xor patterns (SWAPX1/2/4/8).
__device__ __forceinline__ float red_max16(float v) {
    v = fmaxf(v, __int_as_float(__builtin_amdgcn_ds_swizzle(__float_as_int(v), 0x041F)));
    v = fmaxf(v, __int_as_float(__builtin_amdgcn_ds_swizzle(__float_as_int(v), 0x081F)));
    v = fmaxf(v, __int_as_float(__builtin_amdgcn_ds_swizzle(__float_as_int(v), 0x101F)));
    v = fmaxf(v, __int_as_float(__builtin_amdgcn_ds_swizzle(__float_as_int(v), 0x201F)));
    return v;
}
__device__ __forceinline__ float red_sum16(float v) {
    v += __int_as_float(__builtin_amdgcn_ds_swizzle(__float_as_int(v), 0x041F));
    v += __int_as_float(__builtin_amdgcn_ds_swizzle(__float_as_int(v), 0x081F));
    v += __int_as_float(__builtin_amdgcn_ds_swizzle(__float_as_int(v), 0x101F));
    v += __int_as_float(__builtin_amdgcn_ds_swizzle(__float_as_int(v), 0x201F));
    return v;
}

// Load one 16x32 bf16 A/B-style fragment from LDS (two ds_load_b128).
// Lane L -> row rowBase+(L&15); elems 0..7 -> K=k0+(L>>4)*8+e ; 8..15 -> +16.
__device__ __forceinline__ bf16x16 frag_ld(const unsigned short* lds, int rowBase,
                                           int stride, int k0, int lane) {
    const int l16 = lane & 15, half = lane >> 4;
    const unsigned short* p = lds + (size_t)(rowBase + l16) * stride + k0 + half * 8;
    FragU f;
    f.q[0] = *(const u32x4*)(p);
    f.q[1] = *(const u32x4*)(p + 16);
    return f.v;
}

__device__ __forceinline__ f32x8 wmma_bf16(bf16x16 a, bf16x16 b, f32x8 c) {
    return __builtin_amdgcn_wmma_f32_16x16x32_bf16(false, a, false, b, (short)0, c,
                                                   false, false);
}

// ---------------------------------------------------------------------------
// GEMM staging / epilogue helpers (overload on dtype)
// ---------------------------------------------------------------------------
// f32 A: convert while staging (8 contiguous f32 per thread)
__device__ __forceinline__ void stage_A(const float* A, unsigned short* ldsA,
                                        int m0, int k0, int K, int t) {
#pragma unroll
    for (int i = 0; i < 8; ++i) {
        int e = t * 8 + i;
        int r = e >> 5, kk = e & 31;
        ldsA[r * 32 + kk] = f2bf(A[(size_t)(m0 + r) * K + k0 + kk]);
    }
}
// bf16 A: raw 16B async copy into LDS (64x32 tile = 256 chunks, 1/thread)
__device__ __forceinline__ void stage_A(const unsigned short* A, unsigned short* ldsA,
                                        int m0, int k0, int K, int t) {
    int r = t >> 2, ck = t & 3;
    async_copy16(&A[(size_t)(m0 + r) * K + k0 + ck * 8], &ldsA[r * 32 + ck * 8]);
}
__device__ __forceinline__ void store_out(float* C, size_t idx, float v) { C[idx] = v; }
__device__ __forceinline__ void store_out(unsigned short* C, size_t idx, float v) {
    C[idx] = f2bf(v);
}

// ---------------------------------------------------------------------------
// Kernel 1: generic GEMM  C[M,N] = A[M,K] * Bw[K,N] + bias, bf16 WMMA, f32 acc
// block: 256 thr (8 waves, 2x4), block tile 64x128, wave tile 32x32, K step 32
// ---------------------------------------------------------------------------
template <typename TA, typename TC>
__global__ __launch_bounds__(256) void gemm_wmma(const TA* __restrict__ A,
                                                 const float* __restrict__ Bw,
                                                 const float* __restrict__ bias,
                                                 TC* __restrict__ C,
                                                 int M, int N, int K) {
    __shared__ unsigned short ldsA[64 * 32];    // [row][k]
    __shared__ unsigned short ldsB[128 * 32];   // [col][k]  (B transposed)
    const int t    = threadIdx.x;
    const int lane = t & 31, wave = t >> 5;
    const int wm   = wave >> 2, wn = wave & 3;
    const int m0   = blockIdx.y * 64, n0 = blockIdx.x * 128;
    const int l16  = lane & 15, half = lane >> 4;

    f32x8 acc[2][2] = {};

    for (int k0 = 0; k0 < K; k0 += 32) {
        stage_A(A, ldsA, m0, k0, K, t);
        // stage B transposed: coalesced along n, scattered LDS write
#pragma unroll
        for (int i = 0; i < 16; ++i) {
            int e = t * 16 + i;
            int kk = e >> 7, n = e & 127;
            ldsB[n * 32 + kk] = f2bf(Bw[(size_t)(k0 + kk) * N + n0 + n]);
        }
        if (k0 + 32 < K) {   // global_prefetch_b8 of next K-slice
            __builtin_prefetch(&A[(size_t)(m0 + (t & 63)) * K + k0 + 32], 0, 1);
            __builtin_prefetch(&Bw[(size_t)(k0 + 32 + (t & 31)) * N + n0], 0, 1);
        }
        async_wait();
        __syncthreads();

        bf16x16 af0 = frag_ld(ldsA, wm * 32 + 0,  32, 0, lane);
        bf16x16 af1 = frag_ld(ldsA, wm * 32 + 16, 32, 0, lane);
        bf16x16 bf0 = frag_ld(ldsB, wn * 32 + 0,  32, 0, lane);
        bf16x16 bf1 = frag_ld(ldsB, wn * 32 + 16, 32, 0, lane);
        acc[0][0] = wmma_bf16(af0, bf0, acc[0][0]);
        acc[0][1] = wmma_bf16(af0, bf1, acc[0][1]);
        acc[1][0] = wmma_bf16(af1, bf0, acc[1][0]);
        acc[1][1] = wmma_bf16(af1, bf1, acc[1][1]);
        __syncthreads();
    }

    // epilogue: lane owns col n+l16, element r -> row r+8*half
#pragma unroll
    for (int i = 0; i < 2; ++i)
#pragma unroll
        for (int j = 0; j < 2; ++j) {
            int col = n0 + wn * 32 + j * 16 + l16;
            float bv = bias[col];
#pragma unroll
            for (int r = 0; r < 8; ++r) {
                int row = m0 + wm * 32 + i * 16 + r + 8 * half;
                store_out(C, (size_t)row * N + col, acc[i][j][r] + bv);
            }
        }
}

// ---------------------------------------------------------------------------
// Kernel 2: flash-style attention per doubled-head stream (bf16 Q/K/V).
// grid.y = b*16+oh (32 streams), grid.x = query tile of 64 rows.
// block = 128 thr (4 waves, 16 q-rows each). Online softmax over full T.
// PO[hs][t][c] = softmax(q_oh k_ohT * scale) @ V_h   (c in 0..127)
// ---------------------------------------------------------------------------
__global__ __launch_bounds__(128) void diffattn_flash(const unsigned short* __restrict__ Q,
                                                      const unsigned short* __restrict__ Kg,
                                                      const unsigned short* __restrict__ Vg,
                                                      float* __restrict__ PO) {
    __shared__ unsigned short ldsQ[64 * 64];        // [qrow][d]
    __shared__ unsigned short ldsK[32 * 64];        // [key][d]
    __shared__ unsigned short ldsVt[128 * 32];      // [vcol][key]
    __shared__ unsigned short ldsP[4][16 * 32];     // per-wave P transpose scratch

    const int t    = threadIdx.x;
    const int lane = t & 31, wave = t >> 5;
    const int l16  = lane & 15, half = lane >> 4;
    const int hs   = blockIdx.y;            // 0..31
    const int b    = hs >> 4, oh = hs & 15;
    const int h    = oh >> 1;
    const int q0   = blockIdx.x * 64;

    // stage Q tile 64x64 bf16 via async LDS gather (512 x 16B chunks)
#pragma unroll
    for (int i = 0; i < 4; ++i) {
        int e = t * 4 + i;
        int r = e >> 3, ck = e & 7;
        async_copy16(&Q[((size_t)(b * T_) + q0 + r) * HD2_ + oh * 64 + ck * 8],
                     &ldsQ[r * 64 + ck * 8]);
    }
    async_wait();
    __syncthreads();
    const bf16x16 qf0 = frag_ld(ldsQ, wave * 16, 64, 0,  lane);  // d 0..31
    const bf16x16 qf1 = frag_ld(ldsQ, wave * 16, 64, 32, lane);  // d 32..63

    float mrow[8], lrow[8];
    f32x8 O[8] = {};
#pragma unroll
    for (int r = 0; r < 8; ++r) { mrow[r] = -1e30f; lrow[r] = 0.f; }

    for (int j0 = 0; j0 < T_; j0 += 32) {
        __syncthreads();
        // stage K tile 32x64 bf16 via async LDS gather (256 x 16B chunks)
#pragma unroll
        for (int i = 0; i < 2; ++i) {
            int e = t * 2 + i;
            int r = e >> 3, ck = e & 7;
            async_copy16(&Kg[((size_t)(b * T_) + j0 + r) * HD2_ + oh * 64 + ck * 8],
                         &ldsK[r * 64 + ck * 8]);
        }
        // stage V transposed: Vt[c][key]; 2 cols x 2 keys per step,
        // packed with v_perm_b32 into 32-bit LDS stores
#pragma unroll
        for (int i = 0; i < 8; ++i) {
            int e  = t * 8 + i;             // 0..1023 quads
            int cc = (e & 63) * 2;          // column pair base
            int r  = (e >> 6) * 2;          // even key index
            const unsigned short* src =
                &Vg[((size_t)(b * T_) + j0 + r) * HD2_ + h * 128 + cc];
            unsigned int a  = *(const unsigned int*)(src);          // row r
            unsigned int bb = *(const unsigned int*)(src + HD2_);   // row r+1
            *(unsigned int*)&ldsVt[(cc    ) * 32 + r] =
                __builtin_amdgcn_perm(bb, a, 0x05040100u);
            *(unsigned int*)&ldsVt[(cc + 1) * 32 + r] =
                __builtin_amdgcn_perm(bb, a, 0x07060302u);
        }
        async_wait();
        __syncthreads();

        // S = Q * K^T : two 16-key column tiles, K-dim 64 = 2 wmma each
        f32x8 S0 = {}, S1 = {};
        bf16x16 kf;
        kf = frag_ld(ldsK, 0,  64, 0,  lane); S0 = wmma_bf16(qf0, kf, S0);
        kf = frag_ld(ldsK, 0,  64, 32, lane); S0 = wmma_bf16(qf1, kf, S0);
        kf = frag_ld(ldsK, 16, 64, 0,  lane); S1 = wmma_bf16(qf0, kf, S1);
        kf = frag_ld(ldsK, 16, 64, 32, lane); S1 = wmma_bf16(qf1, kf, S1);

        // online softmax; row m = r + 8*half lives in one 16-lane half
        float corr[8];
#pragma unroll
        for (int r = 0; r < 8; ++r) {
            float s0 = S0[r] * SCALE_F, s1 = S1[r] * SCALE_F;
            float tm = red_max16(fmaxf(s0, s1));
            float mn = fmaxf(mrow[r], tm);
            float p0 = __expf(s0 - mn), p1 = __expf(s1 - mn);
            float ts = red_sum16(p0 + p1);
            corr[r] = __expf(mrow[r] - mn);
            lrow[r] = lrow[r] * corr[r] + ts;
            mrow[r] = mn;
            int prow = r + 8 * half;                      // c-frag -> a-frag transpose
            ldsP[wave][prow * 32 + l16]      = f2bf(p0);
            ldsP[wave][prow * 32 + 16 + l16] = f2bf(p1);
        }
#pragma unroll
        for (int n = 0; n < 8; ++n)
#pragma unroll
            for (int r = 0; r < 8; ++r) O[n][r] *= corr[r];

        const bf16x16 pf = frag_ld(ldsP[wave], 0, 32, 0, lane);   // P as A-frag 16x32
#pragma unroll
        for (int n = 0; n < 8; ++n) {
            bf16x16 vb = frag_ld(ldsVt, n * 16, 32, 0, lane);     // V cols n*16..+16
            O[n] = wmma_bf16(pf, vb, O[n]);
        }
    }

    // finalize 1/l and write partial output
#pragma unroll
    for (int n = 0; n < 8; ++n) {
        int c = n * 16 + l16;
#pragma unroll
        for (int r = 0; r < 8; ++r) {
            int qrow = q0 + wave * 16 + r + 8 * half;
            PO[((size_t)hs * T_ + qrow) * 128 + c] = O[n][r] / lrow[r];
        }
    }
}

// ---------------------------------------------------------------------------
// Kernel 3: lambda = exp(sum lq1*lk1) - exp(sum lq2*lk2) + lambda_init
// ---------------------------------------------------------------------------
__global__ void lam_kernel(const float* __restrict__ lq1, const float* __restrict__ lk1,
                           const float* __restrict__ lq2, const float* __restrict__ lk2,
                           float* __restrict__ out) {
    const int t = threadIdx.x;  // 64 threads
    __shared__ float sa[64], sc[64];
    sa[t] = lq1[t] * lk1[t];
    sc[t] = lq2[t] * lk2[t];
    __syncthreads();
    if (t == 0) {
        float s1 = 0.f, s2 = 0.f;
        for (int i = 0; i < 64; ++i) { s1 += sa[i]; s2 += sc[i]; }
        out[0] = __expf(s1) - __expf(s2) + LAMBDA_INIT_F;
    }
}

// ---------------------------------------------------------------------------
// Kernel 4: differential combine (even - lam*odd) + LayerNorm(1024) + gain,
// emits bf16 rows for the output-projection GEMM. One block per (b,t) row.
// ---------------------------------------------------------------------------
__global__ __launch_bounds__(256) void combine_ln(const float* __restrict__ PO,
                                                  const float* __restrict__ lam_p,
                                                  const float* __restrict__ lns,
                                                  const float* __restrict__ lnb,
                                                  unsigned short* __restrict__ out) {
    const int row = blockIdx.x;                  // 0..B*T-1
    const int b = row / T_, tt = row % T_;
    const int t = threadIdx.x;
    const float lam = lam_p[0];

    float x[4];
    float s = 0.f, sq = 0.f;
#pragma unroll
    for (int i = 0; i < 4; ++i) {
        int col = t + i * 256;
        int h = col >> 7, c = col & 127;
        size_t be = (((size_t)(b * 16 + 2 * h))     * T_ + tt) * 128 + c;
        size_t bo = (((size_t)(b * 16 + 2 * h + 1)) * T_ + tt) * 128 + c;
        float v = PO[be] - lam * PO[bo];
        x[i] = v; s += v; sq += v * v;
    }
    __shared__ float rs[256], rq[256];
    rs[t] = s; rq[t] = sq;
    __syncthreads();
    for (int off = 128; off > 0; off >>= 1) {
        if (t < off) { rs[t] += rs[t + off]; rq[t] += rq[t + off]; }
        __syncthreads();
    }
    const float mu  = rs[0] * (1.f / 1024.f);
    const float var = rq[0] * (1.f / 1024.f) - mu * mu;
    const float inv = rsqrtf(var + EPS_F);
    const float og  = 1.f - LAMBDA_INIT_F;
#pragma unroll
    for (int i = 0; i < 4; ++i) {
        int col = t + i * 256;
        out[(size_t)row * 1024 + col] = f2bf(((x[i] - mu) * inv * lns[col] + lnb[col]) * og);
    }
}

// ---------------------------------------------------------------------------
// Launch
// ---------------------------------------------------------------------------
extern "C" void kernel_launch(void* const* d_in, const int* in_sizes, int n_in,
                              void* d_out, int out_size, void* d_ws, size_t ws_size,
                              hipStream_t stream) {
    const float* query = (const float*)d_in[0];
    const float* key   = (const float*)d_in[1];
    const float* value = (const float*)d_in[2];
    const float* Wq = (const float*)d_in[3];  const float* bq = (const float*)d_in[4];
    const float* Wk = (const float*)d_in[5];  const float* bk = (const float*)d_in[6];
    const float* Wv = (const float*)d_in[7];  const float* bv = (const float*)d_in[8];
    const float* lq1 = (const float*)d_in[9];  const float* lk1 = (const float*)d_in[10];
    const float* lq2 = (const float*)d_in[11]; const float* lk2 = (const float*)d_in[12];
    const float* lns = (const float*)d_in[13]; const float* lnb = (const float*)d_in[14];
    const float* Wo = (const float*)d_in[15];  const float* bo = (const float*)d_in[16];
    float* out = (float*)d_out;

    char* ws = (char*)d_ws;
    unsigned short* q_bf = (unsigned short*)(ws);                       // 8 MB [4096,1024] bf16
    unsigned short* k_bf = (unsigned short*)(ws + ((size_t)8  << 20));  // 8 MB
    unsigned short* v_bf = (unsigned short*)(ws + ((size_t)16 << 20));  // 8 MB
    float*          po   = (float*)(ws + ((size_t)24 << 20));           // 32 MB [32,2048,128] f32
    unsigned short* a_bf = (unsigned short*)(ws + ((size_t)56 << 20));  // 8 MB [4096,1024] bf16
    float*          lam  = (float*)(ws + ((size_t)64 << 20));           // 4 B

    const int MT = B_ * T_;  // 4096
    dim3 gblk(256);

    // Q/K/V projections: [4096,512] x [512,1024] -> bf16 activations
    dim3 gproj(HD2_ / 128, MT / 64);
    gemm_wmma<float, unsigned short><<<gproj, gblk, 0, stream>>>(query, Wq, bq, q_bf, MT, HD2_, D_);
    gemm_wmma<float, unsigned short><<<gproj, gblk, 0, stream>>>(key,   Wk, bk, k_bf, MT, HD2_, D_);
    gemm_wmma<float, unsigned short><<<gproj, gblk, 0, stream>>>(value, Wv, bv, v_bf, MT, HD2_, D_);

    // lambda scalar
    lam_kernel<<<1, 64, 0, stream>>>(lq1, lk1, lq2, lk2, lam);

    // flash attention over 32 head-streams x 32 query tiles
    dim3 gattn(T_ / 64, B_ * 2 * H_);
    diffattn_flash<<<gattn, dim3(128), 0, stream>>>(q_bf, k_bf, v_bf, po);

    // differential combine + LayerNorm -> bf16 rows
    combine_ln<<<dim3(MT), dim3(256), 0, stream>>>(po, lam, lns, lnb, a_bf);

    // output projection: [4096,1024](bf16) x [1024,512] -> f32 out
    dim3 gout(M_ / 128, MT / 64);
    gemm_wmma<unsigned short, float><<<gout, gblk, 0, stream>>>(a_bf, Wo, bo, out, MT, M_, HD2_);
}
